// ConvalueAttention_55886114456052
// MI455X (gfx1250) — compile-verified, matrix operand order checked by reading, not run on hardware
//
#include <hip/hip_runtime.h>
#include <hip/hip_bf16.h>

// ConvalueAttention for MI455X (gfx1250), wave32 + WMMA f16->f32.
//
// Pipeline (4 kernels, all on `stream`):
//  1) qk_proj : x @ W_qk^T -> Q (pre-scaled by Dh^-0.5) and K, stored f16
//  2) conv_value : per-token 1D convs 64->40(2ch)->31(4ch) -> v (124, padded 128), f16
//  3) attn : per (b,h,16-row stripe): pass A = WMMA dots + PER-LANE online
//            softmax (single end-of-pass butterfly merge -> minimal DS ops);
//            pass B = recompute dots, write attn (512MB bottleneck) once,
//            accumulate attn@v with WMMA.
//  4) out_proj : out_pre (B,N,992) @ W_out^T + b_out -> out (B,N,64)
//
// d_out = [ out: 4*2048*64 f32 | attn: 4*8*2048*2048 f32 ]

typedef __attribute__((ext_vector_type(16))) _Float16 v16h;
typedef __attribute__((ext_vector_type(8)))  float    v8f;

#define BATCH 4
#define SEQ   2048
#define HEADS 8
#define DH    64
#define VC    124   // conv-value features per head
#define VCP   128   // padded to a multiple of 16 for WMMA tiles

// K index of packed f16 half h (0..15) for this lane, per the CDNA5 16-bit
// A/B matrix VGPR layout (CDNA5 ISA §7.12.2). Consecutive halves
// within each group of 8 are consecutive in K, so per-lane loads vectorize
// to global_load_b128 (confirmed in round-1 disasm).
__device__ __forceinline__ int kofh(int h, int lane) {
  return ((h & 8) << 1) | ((lane & 16) >> 1) | (h & 7);
}

// ---------------------------------------------------------------------------
// Kernel 1: QK projection. One thread per output element (8192 rows x 1024).
// Q is pre-multiplied by scale = Dh^-0.5 = 0.125 so dots need no extra scale.
// ---------------------------------------------------------------------------
__global__ void __launch_bounds__(256)
qk_proj_kernel(const float* __restrict__ x, const float* __restrict__ Wqk,
               _Float16* __restrict__ qws, _Float16* __restrict__ kws) {
  int idx = blockIdx.x * 256 + threadIdx.x;   // exactly 8192*1024 threads
  int o   = idx & 1023;
  int row = idx >> 10;                        // b*SEQ + n
  const float* xr = x + (size_t)row * DH;
  const float* wr = Wqk + (size_t)o * DH;
  float s = 0.f;
#pragma unroll 8
  for (int i = 0; i < DH; ++i) s += xr[i] * wr[i];
  int b = row >> 11, n = row & (SEQ - 1);
  if (o < HEADS * DH) {
    int hd = o >> 6, d = o & 63;
    qws[(((size_t)b * HEADS + hd) * SEQ + n) * DH + d] = (_Float16)(s * 0.125f);
  } else {
    int o2 = o - HEADS * DH;
    int hd = o2 >> 6, d = o2 & 63;
    kws[(((size_t)b * HEADS + hd) * SEQ + n) * DH + d] = (_Float16)s;
  }
}

// ---------------------------------------------------------------------------
// Kernel 2: conv value path. One thread per token: 64 ->(K=25,2ch) 40 ->(K=10,4ch) 31.
// v flattened channel-major (c*31 + t) = 124 features, zero-padded to 128, f16.
// ---------------------------------------------------------------------------
__global__ void __launch_bounds__(64)
conv_value_kernel(const float* __restrict__ x,
                  const float* __restrict__ w1, const float* __restrict__ b1,
                  const float* __restrict__ w2, const float* __restrict__ b2,
                  _Float16* __restrict__ vws) {
  int row = blockIdx.x * 64 + threadIdx.x;    // 0..8191
  if (row >= BATCH * SEQ) return;
  const float* xr = x + (size_t)row * DH;

  float h1[2][40];
#pragma unroll
  for (int c = 0; c < 2; ++c)
    for (int t = 0; t < 40; ++t) {
      float s = b1[c];
#pragma unroll
      for (int k = 0; k < 25; ++k) s += xr[t + k] * w1[c * 25 + k];
      h1[c][t] = s;
    }

  _Float16* vr = vws + (size_t)row * VCP;
#pragma unroll
  for (int c = 0; c < 4; ++c)
    for (int t = 0; t < 31; ++t) {
      float s = b2[c];
#pragma unroll
      for (int ci = 0; ci < 2; ++ci)
#pragma unroll
        for (int k = 0; k < 10; ++k) s += h1[ci][t + k] * w2[(c * 2 + ci) * 10 + k];
      vr[c * 31 + t] = (_Float16)s;
    }
  vr[124] = (_Float16)0.f; vr[125] = (_Float16)0.f;
  vr[126] = (_Float16)0.f; vr[127] = (_Float16)0.f;
}

// ---------------------------------------------------------------------------
// Kernel 3: attention. One wave32 per (b, h, 16-query stripe) -> 4096 waves.
// ---------------------------------------------------------------------------
__global__ void __launch_bounds__(32)
attn_kernel(const _Float16* __restrict__ qws, const _Float16* __restrict__ kws,
            const _Float16* __restrict__ vws, float* __restrict__ attn_out,
            float* __restrict__ outpre) {
  const int wg   = blockIdx.x;      // 0..4095
  const int mt   = wg & 127;        // query tile within (b,h)
  const int bh   = wg >> 7;         // b*HEADS + h
  const int head = bh & (HEADS - 1);
  const int b    = bh >> 3;
  const int lane = threadIdx.x;
  const int lm   = lane & 15;
  const int lh   = lane >> 4;       // 0: rows 0-7, 1: rows 8-15 of the C tile

  const _Float16* qrow  = qws + ((size_t)bh * SEQ + mt * 16 + lm) * DH;
  const _Float16* kbase = kws + (size_t)bh * SEQ * DH;
  const _Float16* vbase = vws + (size_t)b * SEQ * VCP;

  // A-matrix tiles of Q: 16x64 split into two 16x32 f16 tiles.
  v16h aq0, aq1;
#pragma unroll
  for (int t = 0; t < 16; ++t) {
    int kk = kofh(t, lane);
    aq0[t] = qrow[kk];
    aq1[t] = qrow[32 + kk];
  }

  // ---- Pass A: per-lane online softmax stats (no cross-lane work in the
  // hot loop: each lane privately covers one column per tile per row). ----
  float mrun[8], srun[8];
#pragma unroll
  for (int r = 0; r < 8; ++r) { mrun[r] = -1e30f; srun[r] = 0.f; }

  for (int j = 0; j < 128; ++j) {
    const _Float16* krow = kbase + (size_t)(j * 16 + lm) * DH;
    v16h bk0, bk1;
#pragma unroll
    for (int t = 0; t < 16; ++t) {
      int kk = kofh(t, lane);
      bk0[t] = krow[kk];        // B = K^T : B[kk][n] = k[n, kk]
      bk1[t] = krow[32 + kk];
    }
    v8f c = {};
    c = __builtin_amdgcn_wmma_f32_16x16x32_f16(false, aq0, false, bk0, (short)0, c, false, false);
    c = __builtin_amdgcn_wmma_f32_16x16x32_f16(false, aq1, false, bk1, (short)0, c, false, false);
#pragma unroll
    for (int r = 0; r < 8; ++r) {
      float d  = c[r];
      float nm = fmaxf(mrun[r], d);
      srun[r]  = srun[r] * __expf(mrun[r] - nm) + __expf(d - nm);
      mrun[r]  = nm;
    }
  }

  // Single butterfly merge of the 16 per-lane (m,s) states per half-wave.
  // Online-softmax combine is associative: (m,s)+(m',s') ->
  //   (max, s*e^{m-max} + s'*e^{m'-max}).  64 DS ops total vs 8K before.
#pragma unroll
  for (int mask = 1; mask <= 8; mask <<= 1) {
#pragma unroll
    for (int r = 0; r < 8; ++r) {
      float mo = __shfl_xor(mrun[r], mask, 32);
      float so = __shfl_xor(srun[r], mask, 32);
      float nm = fmaxf(mrun[r], mo);
      srun[r]  = srun[r] * __expf(mrun[r] - nm) + so * __expf(mo - nm);
      mrun[r]  = nm;
    }
  }
  float rinv[8];
#pragma unroll
  for (int r = 0; r < 8; ++r) rinv[r] = 1.f / srun[r];

  // ---- Pass B: recompute dots, emit attn, accumulate attn @ v ----
  __shared__ _Float16 lp[16][32];   // D-tile -> A-tile relayout staging
  v8f zero = {};
  v8f acc[8];
#pragma unroll
  for (int ct = 0; ct < 8; ++ct) acc[ct] = zero;

  float* attnbase = attn_out + ((size_t)bh * SEQ + mt * 16) * SEQ;

  for (int j2 = 0; j2 < 64; ++j2) {      // 32 key columns per iteration
#pragma unroll
    for (int sub = 0; sub < 2; ++sub) {
      int j = j2 * 2 + sub;
      const _Float16* krow = kbase + (size_t)(j * 16 + lm) * DH;
      v16h bk0, bk1;
#pragma unroll
      for (int t = 0; t < 16; ++t) {
        int kk = kofh(t, lane);
        bk0[t] = krow[kk];
        bk1[t] = krow[32 + kk];
      }
      v8f c = {};
      c = __builtin_amdgcn_wmma_f32_16x16x32_f16(false, aq0, false, bk0, (short)0, c, false, false);
      c = __builtin_amdgcn_wmma_f32_16x16x32_f16(false, aq1, false, bk1, (short)0, c, false, false);
#pragma unroll
      for (int r = 0; r < 8; ++r) {
        float p  = __expf(c[r] - mrun[r]) * rinv[r];
        int row  = r + 8 * lh;
        attnbase[(size_t)row * SEQ + j * 16 + lm] = p;   // final attn output
        lp[row][sub * 16 + lm] = (_Float16)p;
      }
    }
    __syncthreads();   // single-wave WG: cheap; orders LDS write -> read

    v16h ap;           // A-tile of probabilities: 16 rows x 32 keys
#pragma unroll
    for (int t = 0; t < 16; ++t) ap[t] = lp[lm][kofh(t, lane)];

#pragma unroll
    for (int ct = 0; ct < 8; ++ct) {     // 8 column tiles cover VCP=128
      v16h bv;
#pragma unroll
      for (int t = 0; t < 16; ++t)
        bv[t] = vbase[(size_t)(j2 * 32 + kofh(t, lane)) * VCP + ct * 16 + lm];
      acc[ct] = __builtin_amdgcn_wmma_f32_16x16x32_f16(false, ap, false, bv, (short)0, acc[ct], false, false);
    }
    __syncthreads();
  }

  // Store out_pre in (b, n, head, c) order so kernel 4 sees contiguous 992.
#pragma unroll
  for (int ct = 0; ct < 8; ++ct) {
    int cc = ct * 16 + lm;
    if (cc < VC) {
#pragma unroll
      for (int r = 0; r < 8; ++r) {
        int n = mt * 16 + r + 8 * lh;
        outpre[(((size_t)b * SEQ + n) * HEADS + head) * VC + cc] = acc[ct][r];
      }
    }
  }
}

// ---------------------------------------------------------------------------
// Kernel 4: output projection. out[b,n,d] = out_pre[b,n,:] . W_out[d,:] + b_out[d]
// ---------------------------------------------------------------------------
__global__ void __launch_bounds__(256)
out_proj_kernel(const float* __restrict__ outpre, const float* __restrict__ Wout,
                const float* __restrict__ bout, float* __restrict__ out) {
  int idx = blockIdx.x * 256 + threadIdx.x;   // exactly 4*2048*64 threads
  int d   = idx & 63;
  int row = idx >> 6;                         // b*SEQ + n
  const float* pr = outpre + (size_t)row * (HEADS * VC);
  const float* wr = Wout + (size_t)d * (HEADS * VC);
  float s = bout[d];
#pragma unroll 8
  for (int k = 0; k < HEADS * VC; ++k) s += pr[k] * wr[k];
  out[idx] = s;
}

// ---------------------------------------------------------------------------
extern "C" void kernel_launch(void* const* d_in, const int* in_sizes, int n_in,
                              void* d_out, int out_size, void* d_ws, size_t ws_size,
                              hipStream_t stream) {
  const float* x    = (const float*)d_in[0];
  const float* Wqk  = (const float*)d_in[1];
  const float* w1   = (const float*)d_in[2];
  const float* b1   = (const float*)d_in[3];
  const float* w2   = (const float*)d_in[4];
  const float* b2   = (const float*)d_in[5];
  const float* Wout = (const float*)d_in[6];
  const float* bout = (const float*)d_in[7];

  float* out  = (float*)d_out;                                  // (B,N,64)
  float* attn = out + (size_t)BATCH * SEQ * DH;                 // (B,H,N,N)

  // Workspace layout (~51.4 MB total):
  char* ws = (char*)d_ws;
  _Float16* qws    = (_Float16*)(ws);                           // 8.0 MB f16 Q
  _Float16* kws    = (_Float16*)(ws + 8388608);                 // 8.0 MB f16 K
  _Float16* vws    = (_Float16*)(ws + 16777216);                // 2.0 MB f16 V (padded 128)
  float*    outpre = (float*)   (ws + 18874368);                // 32.5 MB f32 (B,N,8,124)

  qk_proj_kernel   <<<32768, 256, 0, stream>>>(x, Wqk, qws, kws);
  conv_value_kernel<<<128,    64, 0, stream>>>(x, w1, b1, w2, b2, vws);
  attn_kernel      <<<BATCH * HEADS * (SEQ / 16), 32, 0, stream>>>(qws, kws, vws, attn, outpre);
  out_proj_kernel  <<<(BATCH * SEQ * DH) / 256, 256, 0, stream>>>(outpre, Wout, bout, out);
}
